// PKGNN_5566277616234
// MI455X (gfx1250) — matrix operand order repaired and checked by reading, not compiled, not run on hardware
//
#include <hip/hip_runtime.h>
#include <math.h>

// Problem constants (match reference)
#define N_NODES 20000
#define N_EDGES 320000
#define F_INP   128
#define EDIM    32
#define HEADS   4
#define CH      32          // per-head channels
#define HCN     128         // HEADS*CH
#define LAYERS  2
#define GRAPHS  64
#define DNN     256
#define DNN2    128
#define NCLS    4

typedef __attribute__((ext_vector_type(16))) __bf16 v16bf;
typedef __attribute__((ext_vector_type(8)))  __bf16 v8bf;
typedef __attribute__((ext_vector_type(8)))  float  v8f;

// ---------------------------------------------------------------- utilities
__global__ void fill_f32(float* __restrict__ p, float v, int n) {
  int i = blockIdx.x * blockDim.x + threadIdx.x;
  if (i < n) p[i] = v;
}

__device__ inline float atomicMaxF(float* addr, float value) {
  // sign-punned float max: correct for any mix, init to -inf
  if (value >= 0.0f)
    return __int_as_float(atomicMax((int*)addr, __float_as_int(value)));
  return __uint_as_float(atomicMin((unsigned int*)addr, __float_as_uint(value)));
}

__device__ inline void store_bf16x4(__bf16* dst, float a, float b, float c, float d) {
  union { __bf16 h[4]; unsigned long long u; } pk;
  pk.h[0] = (__bf16)a; pk.h[1] = (__bf16)b; pk.h[2] = (__bf16)c; pk.h[3] = (__bf16)d;
  *reinterpret_cast<unsigned long long*>(dst) = pk.u;   // 8B-aligned LDS store
}

// ---------------------------------------------------------------- WMMA GEMM
// Y[M,NO] = act( X[M,K] @ W[K,NO] + bias )   (bias may be null, act: 0/1=relu)
// Assumptions (hold for every GEMM in this net): M % 16 == 0, K % 32 == 0,
// NO % 4 == 0, NO <= 256.
// Block: 256 threads = 8 waves. Block tile: 16 rows x NO cols.
// Each wave owns n-tiles {wave, wave+8} (NO <= 256 -> at most 2 tiles/wave).
__global__ __launch_bounds__(256)
void gemm_bias_act(const float* __restrict__ X, const float* __restrict__ W,
                   const float* __restrict__ bias, float* __restrict__ Y,
                   int M, int K, int NO, int act) {
  __shared__ __bf16 sA[16 * 32];           // A tile, row-major [16][32]   (1 KB)
  __shared__ __bf16 sWf[16 * 32 * 16];     // B tile, fragment-order
                                           // [ntile][lane][16 elems]     (16 KB)
  const int tid  = threadIdx.x;
  const int lane = tid & 31;
  const int wave = tid >> 5;
  const int m0   = blockIdx.x * 16;
  const int ntiles = (NO + 15) >> 4;

  v8f acc0 = {0.f,0.f,0.f,0.f,0.f,0.f,0.f,0.f};
  v8f acc1 = {0.f,0.f,0.f,0.f,0.f,0.f,0.f,0.f};
  const int nt0 = wave;
  const int nt1 = wave + 8;

  for (int k0 = 0; k0 < K; k0 += 32) {
    // ---- A tile: 16x32 floats = 128 float4 groups, threads 0..127
    if (tid < 128) {
      const int r  = tid >> 3;            // row 0..15
      const int c4 = (tid & 7) << 2;      // col 0,4,...,28
      const float4 f = *reinterpret_cast<const float4*>(
          X + (long)(m0 + r) * K + k0 + c4);
      store_bf16x4(&sA[(r << 5) + c4], f.x, f.y, f.z, f.w);
    }
    // ---- W tile, staged directly in WMMA B-fragment order.
    // Group g = (k-quad r4, column col): 4 coalesced global b32 loads,
    // one 8-byte LDS store at [ntile][lane][elem_base].
    for (int g = tid; g < 8 * NO; g += 256) {
      const int col = g % NO;
      const int r4  = g / NO;             // 0..7 -> k rows 4*r4 .. 4*r4+3
      const float* wp = W + (long)(k0 + (r4 << 2)) * NO + col;
      const float f0 = wp[0];
      const float f1 = wp[NO];
      const float f2 = wp[2 * NO];
      const float f3 = wp[3 * NO];
      const int l   = (col & 15) + ((r4 >= 4) ? 16 : 0);  // B: lane = N, +16 for K>=16
      const int ntb = col >> 4;
      store_bf16x4(&sWf[(((ntb << 5) + l) << 4) + ((r4 & 3) << 2)], f0, f1, f2, f3);
    }
    __syncthreads();

    // ---- A fragment (16-bit A 16x32, contiguous k = 16*(lane/16) + i): two contiguous 16B runs
    v16bf afrag;
    {
      const int row = lane & 15;
      const int kb  = (lane >> 4) << 4;   // 0 or 16
      const v8bf lo = *reinterpret_cast<const v8bf*>(&sA[(row << 5) + kb]);
      const v8bf hi = *reinterpret_cast<const v8bf*>(&sA[(row << 5) + 8 + kb]);
#pragma unroll
      for (int i = 0; i < 8; ++i) { afrag[i] = lo[i]; afrag[8 + i] = hi[i]; }
    }

    // ---- B fragments: one 32B contiguous LDS read each
    if (nt0 < ntiles) {
      const v16bf bfrag =
          *reinterpret_cast<const v16bf*>(&sWf[((nt0 << 5) + lane) << 4]);
      acc0 = __builtin_amdgcn_wmma_f32_16x16x32_bf16(
          false, afrag, false, bfrag, (short)0, acc0, false, false);
    }
    if (nt1 < ntiles) {
      const v16bf bfrag =
          *reinterpret_cast<const v16bf*>(&sWf[((nt1 << 5) + lane) << 4]);
      acc1 = __builtin_amdgcn_wmma_f32_16x16x32_bf16(
          false, afrag, false, bfrag, (short)0, acc1, false, false);
    }
    __syncthreads();
  }

  // ---- store: lane 0-15 -> N=lane, VGPR r -> M=r (+8 for upper half-lanes)
  const int mbase = m0 + ((lane >> 4) << 3);
  if (nt0 < ntiles) {
    const int col = (nt0 << 4) + (lane & 15);
    if (col < NO) {
      const float bv = bias ? bias[col] : 0.0f;
#pragma unroll
      for (int r = 0; r < 8; ++r) {
        float val = acc0[r] + bv;
        if (act) val = fmaxf(val, 0.0f);
        Y[(long)(mbase + r) * NO + col] = val;
      }
    }
  }
  if (nt1 < ntiles) {
    const int col = (nt1 << 4) + (lane & 15);
    if (col < NO) {
      const float bv = bias ? bias[col] : 0.0f;
#pragma unroll
      for (int r = 0; r < 8; ++r) {
        float val = acc1[r] + bv;
        if (act) val = fmaxf(val, 0.0f);
        Y[(long)(mbase + r) * NO + col] = val;
      }
    }
  }
}

// ---------------------------------------------------------------- edge phase
__global__ __launch_bounds__(256)
void edge_scores(const float* __restrict__ q, const float* __restrict__ k,
                 const float* __restrict__ e, const int* __restrict__ src,
                 const int* __restrict__ dst, float* __restrict__ scores,
                 float* __restrict__ mbuf, int E) {
  int t = blockIdx.x * blockDim.x + threadIdx.x;
  if (t >= E * HEADS) return;
  int ed = t >> 2, h = t & 3;
  int s = src[ed], d = dst[ed];
  const float4* qp = reinterpret_cast<const float4*>(q + (long)d * HCN + h * CH);
  const float4* kp = reinterpret_cast<const float4*>(k + (long)s * HCN + h * CH);
  const float4* ep = reinterpret_cast<const float4*>(e + (long)ed * HCN + h * CH);
  float acc = 0.0f;
#pragma unroll
  for (int c = 0; c < CH / 4; ++c) {
    const float4 qq = qp[c], kv = kp[c], ev = ep[c];
    acc += qq.x * (kv.x + ev.x) + qq.y * (kv.y + ev.y) +
           qq.z * (kv.z + ev.z) + qq.w * (kv.w + ev.w);
  }
  acc *= 0.17677669529663687f;   // 1/sqrt(32)
  scores[t] = acc;
  atomicMaxF(&mbuf[(long)d * HEADS + h], acc);
}

__global__ __launch_bounds__(256)
void edge_exp(const float* __restrict__ scores, const float* __restrict__ mbuf,
              const int* __restrict__ dst, float* __restrict__ ex,
              float* __restrict__ den, int E) {
  int t = blockIdx.x * blockDim.x + threadIdx.x;
  if (t >= E * HEADS) return;
  int ed = t >> 2, h = t & 3;
  int d = dst[ed];
  float v = __expf(scores[t] - mbuf[(long)d * HEADS + h]);
  ex[t] = v;
  atomicAdd(&den[(long)d * HEADS + h], v);
}

// one thread per (edge, 4 channels): float4 gathers, 4 scatter atomics
__global__ __launch_bounds__(256)
void edge_aggregate(const float* __restrict__ v, const float* __restrict__ e,
                    const float* __restrict__ ex, const float* __restrict__ den,
                    const int* __restrict__ src, const int* __restrict__ dst,
                    float* __restrict__ agg, int E) {
  long t = (long)blockIdx.x * blockDim.x + threadIdx.x;
  if (t >= (long)E * (HCN / 4)) return;
  int ed = (int)(t >> 5);
  int c4 = (int)(t & 31) << 2;          // 0,4,...,124
  int h  = c4 >> 5;
  int s = src[ed], d = dst[ed];
  float alpha = ex[(long)ed * HEADS + h] / (den[(long)d * HEADS + h] + 1e-16f);
  const float4 vv = *reinterpret_cast<const float4*>(v + (long)s * HCN + c4);
  const float4 ev = *reinterpret_cast<const float4*>(e + (long)ed * HCN + c4);
  float* ap = agg + (long)d * HCN + c4;
  atomicAdd(ap + 0, (vv.x + ev.x) * alpha);
  atomicAdd(ap + 1, (vv.y + ev.y) * alpha);
  atomicAdd(ap + 2, (vv.z + ev.z) * alpha);
  atomicAdd(ap + 3, (vv.w + ev.w) * alpha);
}

// gated skip: beta = sigmoid([out, xr, out-xr] @ Wb); h = beta*xr+(1-beta)*out
// one wave32 per node; each lane owns 4 channels (stride-32, coalesced)
__global__ __launch_bounds__(256)
void beta_combine(const float* __restrict__ agg, const float* __restrict__ xr,
                  const float* __restrict__ Wb, float* __restrict__ out, int n) {
  int node = blockIdx.x * 8 + (threadIdx.x >> 5);
  int lane = threadIdx.x & 31;
  if (node >= n) return;
  float av[4], xv[4], s = 0.0f;
#pragma unroll
  for (int j = 0; j < 4; ++j) {
    int c = lane + j * 32;
    float a = agg[(long)node * HCN + c];
    float x = xr[(long)node * HCN + c];
    av[j] = a; xv[j] = x;
    s += Wb[c] * a + Wb[HCN + c] * x + Wb[2 * HCN + c] * (a - x);
  }
#pragma unroll
  for (int off = 16; off > 0; off >>= 1) s += __shfl_xor(s, off, 32);
  float beta = 1.0f / (1.0f + __expf(-s));
#pragma unroll
  for (int j = 0; j < 4; ++j) {
    int c = lane + j * 32;
    out[(long)node * HCN + c] = beta * xv[j] + (1.0f - beta) * av[j];
  }
}

// ---------------------------------------------------------------- batchnorm
__global__ __launch_bounds__(256)
void bn_stats(const float* __restrict__ h, float* __restrict__ stats, int n) {
  __shared__ float ls[2 * CH];
  if (threadIdx.x < 2 * CH) ls[threadIdx.x] = 0.0f;
  __syncthreads();
  long total = (long)n * CH;
  for (long i = (long)blockIdx.x * blockDim.x + threadIdx.x; i < total;
       i += (long)gridDim.x * blockDim.x) {
    int c = (int)(i & 31);
    float v = h[i];
    atomicAdd(&ls[c], v);              // LDS atomics (ds_add_f32)
    atomicAdd(&ls[CH + c], v * v);
  }
  __syncthreads();
  if (threadIdx.x < 2 * CH) atomicAdd(&stats[threadIdx.x], ls[threadIdx.x]);
}

__global__ __launch_bounds__(256)
void bn_apply(float* __restrict__ h, const float* __restrict__ stats,
              const float* __restrict__ g, const float* __restrict__ b, int n) {
  long i = (long)blockIdx.x * blockDim.x + threadIdx.x;
  if (i >= (long)n * CH) return;
  int c = (int)(i & 31);
  float invn = 1.0f / (float)n;
  float mu = stats[c] * invn;
  float var = stats[CH + c] * invn - mu * mu;
  h[i] = (h[i] - mu) * rsqrtf(var + 1e-5f) * g[c] + b[c];
}

// ---------------------------------------------------------------- pooling
__global__ __launch_bounds__(256)
void pool_accum(const float* __restrict__ h, const int* __restrict__ batch,
                float* __restrict__ mx, float* __restrict__ sm,
                float* __restrict__ cnt, int n) {
  long i = (long)blockIdx.x * blockDim.x + threadIdx.x;
  if (i >= (long)n * CH) return;
  int node = (int)(i >> 5), c = (int)(i & 31);
  int g = batch[node];
  float v = h[i];
  atomicMaxF(&mx[g * CH + c], v);
  atomicAdd(&sm[g * CH + c], v);
  if (c == 0) atomicAdd(&cnt[g], 1.0f);
}

__global__ __launch_bounds__(256)
void pool_finalize(float* __restrict__ rep, const float* __restrict__ mx,
                   const float* __restrict__ sm, const float* __restrict__ cnt) {
  int i = blockIdx.x * blockDim.x + threadIdx.x;
  if (i >= GRAPHS * CH) return;
  int g = i >> 5, c = i & 31;
  float m = mx[i];
  if (!(m > -1e30f)) m = 0.0f;              // non-finite -> 0
  float count = fmaxf(cnt[g], 1.0f);
  rep[g * 2 * CH + c]      += m;
  rep[g * 2 * CH + CH + c] += sm[i] / count;
}

// ---------------------------------------------------------------- driver
extern "C" void kernel_launch(void* const* d_in, const int* in_sizes, int n_in,
                              void* d_out, int out_size, void* d_ws, size_t ws_size,
                              hipStream_t stream) {
  (void)in_sizes; (void)n_in; (void)out_size; (void)ws_size;
  const float* x     = (const float*)d_in[0];
  const int*   ei    = (const int*)d_in[1];
  const int*   src   = ei;
  const int*   dst   = ei + N_EDGES;
  const float* ea    = (const float*)d_in[2];
  const int*   batch = (const int*)d_in[3];
  const float* c1_Wq = (const float*)d_in[4];  const float* c1_bq = (const float*)d_in[5];
  const float* c1_Wk = (const float*)d_in[6];  const float* c1_bk = (const float*)d_in[7];
  const float* c1_Wv = (const float*)d_in[8];  const float* c1_bv = (const float*)d_in[9];
  const float* c1_We = (const float*)d_in[10];
  const float* c1_Ws = (const float*)d_in[11]; const float* c1_bs = (const float*)d_in[12];
  const float* c1_Wb = (const float*)d_in[13];
  const float* t1_W  = (const float*)d_in[14]; const float* t1_b  = (const float*)d_in[15];
  const float* bn1_g = (const float*)d_in[16]; const float* bn1_b = (const float*)d_in[17];
  const float* Wq_s  = (const float*)d_in[18]; const float* bq_s  = (const float*)d_in[19];
  const float* Wk_s  = (const float*)d_in[20]; const float* bk_s  = (const float*)d_in[21];
  const float* Wv_s  = (const float*)d_in[22]; const float* bv_s  = (const float*)d_in[23];
  const float* We_s  = (const float*)d_in[24];
  const float* Ws_s  = (const float*)d_in[25]; const float* bs_s  = (const float*)d_in[26];
  const float* Wb_s  = (const float*)d_in[27];
  const float* tW_s  = (const float*)d_in[28]; const float* tb_s  = (const float*)d_in[29];
  const float* bng_s = (const float*)d_in[30]; const float* bnb_s = (const float*)d_in[31];
  const float* l1_W  = (const float*)d_in[32]; const float* l1_b  = (const float*)d_in[33];
  const float* l2_W  = (const float*)d_in[34]; const float* l2_b  = (const float*)d_in[35];
  const float* l3_W  = (const float*)d_in[36]; const float* l3_b  = (const float*)d_in[37];

  // workspace carve-out (floats)
  float* ws = (float*)d_ws;
  size_t off = 0;
  auto carve = [&](size_t nel) { float* p = ws + off; off += nel; return p; };
  float* eemb   = carve((size_t)N_EDGES * HCN);
  float* q      = carve((size_t)N_NODES * HCN);
  float* kk     = carve((size_t)N_NODES * HCN);
  float* vv     = carve((size_t)N_NODES * HCN);
  float* agg    = carve((size_t)N_NODES * HCN);
  float* xr     = carve((size_t)N_NODES * HCN);
  float* hb     = carve((size_t)N_NODES * HCN);
  float* h      = carve((size_t)N_NODES * CH);
  float* scores = carve((size_t)N_EDGES * HEADS);
  float* ex     = carve((size_t)N_EDGES * HEADS);
  float* mb     = carve((size_t)N_NODES * HEADS);
  float* den    = carve((size_t)N_NODES * HEADS);
  float* stats  = carve(2 * CH);
  float* pmx    = carve(GRAPHS * CH);
  float* psm    = carve(GRAPHS * CH);
  float* pcnt   = carve(GRAPHS);
  float* rep    = carve(GRAPHS * 2 * CH);
  float* z1     = carve((size_t)GRAPHS * DNN);
  float* z2     = carve((size_t)GRAPHS * DNN2);

  const int gN  = N_NODES / 16;
  const int gE  = N_EDGES / 16;
  const int gEH = (N_EDGES * HEADS + 255) / 256;
  const int gEC = (int)(((long)N_EDGES * (HCN / 4) + 255) / 256);
  const int gNC = ((N_NODES * CH) + 255) / 256;

  auto conv = [&](const float* xin, int fin,
                  const float* Wq_, const float* bq_, const float* Wk_, const float* bk_,
                  const float* Wv_, const float* bv_, const float* We_,
                  const float* Wsk_, const float* bsk_, const float* Wbeta_,
                  const float* tW_, const float* tb_, const float* bg_, const float* bb_,
                  float* hout) {
    gemm_bias_act<<<gN, 256, 0, stream>>>(xin, Wq_,  bq_,  q,    N_NODES, fin, HCN, 0);
    gemm_bias_act<<<gN, 256, 0, stream>>>(xin, Wk_,  bk_,  kk,   N_NODES, fin, HCN, 0);
    gemm_bias_act<<<gN, 256, 0, stream>>>(xin, Wv_,  bv_,  vv,   N_NODES, fin, HCN, 0);
    gemm_bias_act<<<gN, 256, 0, stream>>>(xin, Wsk_, bsk_, xr,   N_NODES, fin, HCN, 0);
    gemm_bias_act<<<gE, 256, 0, stream>>>(ea,  We_,  nullptr, eemb, N_EDGES, EDIM, HCN, 0);
    fill_f32<<<(N_NODES * HEADS + 255) / 256, 256, 0, stream>>>(mb, -INFINITY, N_NODES * HEADS);
    fill_f32<<<(N_NODES * HEADS + 255) / 256, 256, 0, stream>>>(den, 0.0f, N_NODES * HEADS);
    fill_f32<<<(N_NODES * HCN + 255) / 256, 256, 0, stream>>>(agg, 0.0f, N_NODES * HCN);
    edge_scores<<<gEH, 256, 0, stream>>>(q, kk, eemb, src, dst, scores, mb, N_EDGES);
    edge_exp<<<gEH, 256, 0, stream>>>(scores, mb, dst, ex, den, N_EDGES);
    edge_aggregate<<<gEC, 256, 0, stream>>>(vv, eemb, ex, den, src, dst, agg, N_EDGES);
    beta_combine<<<(N_NODES + 7) / 8, 256, 0, stream>>>(agg, xr, Wbeta_, hb, N_NODES);
    gemm_bias_act<<<gN, 256, 0, stream>>>(hb, tW_, tb_, hout, N_NODES, HCN, CH, 1);
    fill_f32<<<1, 64, 0, stream>>>(stats, 0.0f, 2 * CH);
    bn_stats<<<2048, 256, 0, stream>>>(hout, stats, N_NODES);
    bn_apply<<<gNC, 256, 0, stream>>>(hout, stats, bg_, bb_, N_NODES);
  };

  // conv1 (F_IN -> HCN -> C)
  conv(x, F_INP, c1_Wq, c1_bq, c1_Wk, c1_bk, c1_Wv, c1_bv, c1_We,
       c1_Ws, c1_bs, c1_Wb, t1_W, t1_b, bn1_g, bn1_b, h);

  fill_f32<<<(GRAPHS * 2 * CH + 255) / 256, 256, 0, stream>>>(rep, 0.0f, GRAPHS * 2 * CH);

  for (int i = 0; i < LAYERS; ++i) {
    conv(h, CH,
         Wq_s + (size_t)i * CH * HCN, bq_s + (size_t)i * HCN,
         Wk_s + (size_t)i * CH * HCN, bk_s + (size_t)i * HCN,
         Wv_s + (size_t)i * CH * HCN, bv_s + (size_t)i * HCN,
         We_s + (size_t)i * EDIM * HCN,
         Ws_s + (size_t)i * CH * HCN, bs_s + (size_t)i * HCN,
         Wb_s + (size_t)i * 3 * HCN,
         tW_s + (size_t)i * HCN * CH, tb_s + (size_t)i * CH,
         bng_s + (size_t)i * CH, bnb_s + (size_t)i * CH, h);
    fill_f32<<<(GRAPHS * CH + 255) / 256, 256, 0, stream>>>(pmx, -INFINITY, GRAPHS * CH);
    fill_f32<<<(GRAPHS * CH + 255) / 256, 256, 0, stream>>>(psm, 0.0f, GRAPHS * CH);
    fill_f32<<<1, 64, 0, stream>>>(pcnt, 0.0f, GRAPHS);
    pool_accum<<<gNC, 256, 0, stream>>>(h, batch, pmx, psm, pcnt, N_NODES);
    pool_finalize<<<(GRAPHS * CH + 255) / 256, 256, 0, stream>>>(rep, pmx, psm, pcnt);
  }

  // MLP head (all WMMA GEMMs)
  const int gG = GRAPHS / 16;
  gemm_bias_act<<<gG, 256, 0, stream>>>(rep, l1_W, l1_b, z1, GRAPHS, 2 * CH, DNN, 1);
  gemm_bias_act<<<gG, 256, 0, stream>>>(z1, l2_W, l2_b, z2, GRAPHS, DNN, DNN2, 1);
  gemm_bias_act<<<gG, 256, 0, stream>>>(z2, l3_W, l3_b, (float*)d_out, GRAPHS, DNN2, NCLS, 0);
}
